// QLoRALinear_6992206758367
// MI455X (gfx1250) — matrix operand order repaired and checked by reading, not compiled, not run on hardware
//
#include <hip/hip_runtime.h>
#include <hip/hip_fp16.h>

// Problem dims fixed by the reference: x[4,2048,4096], W[4096,4096] 4-bit, r=16
#define MDIM 8192   // B*S
#define KDIM 4096   // IN
#define NDIM 4096   // OUT
#define SCALING 2.0f

#define BM 128
#define BN 128
#define BK 32
#define KPAD 40     // 80-byte LDS row stride -> conflict-free 16B ds reads

typedef __attribute__((ext_vector_type(16))) _Float16 v16h;
typedef __attribute__((ext_vector_type(8)))  _Float16 v8h;
typedef __attribute__((ext_vector_type(4)))  _Float16 v4h;
typedef __attribute__((ext_vector_type(2)))  _Float16 v2h;
typedef __attribute__((ext_vector_type(8)))  float    v8f;
typedef __attribute__((ext_vector_type(4)))  float    v4f;
typedef __attribute__((ext_vector_type(4)))  int      i32x4;

#if defined(__AMDGCN__) && __has_builtin(__builtin_amdgcn_global_load_async_to_lds_b128)
#define HAVE_ASYNC_LDS 1
#else
#define HAVE_ASYNC_LDS 0
#endif

#if HAVE_ASYNC_LDS && __has_builtin(__builtin_amdgcn_s_wait_asynccnt)
#define WAIT_ASYNC(n) __builtin_amdgcn_s_wait_asynccnt(n)
#else
#define WAIT_ASYNC(n)
#endif

// 16-byte global -> LDS copy: async DMA path on CDNA5, sync fallback otherwise.
__device__ __forceinline__ void g2l_16B(const _Float16* g, _Float16* l) {
#if HAVE_ASYNC_LDS
  __builtin_amdgcn_global_load_async_to_lds_b128(
      (__attribute__((address_space(1))) i32x4*)(g),
      (__attribute__((address_space(3))) i32x4*)(l), 0, 0);
#else
  *(v8h*)l = *(const v8h*)g;
#endif
}

// ---------------- kernel 1: fp32 x -> fp16 Xh --------------------------------
__global__ __launch_bounds__(256)
void k_cvt_x(const float* __restrict__ x, _Float16* __restrict__ xh) {
  size_t i = (size_t)blockIdx.x * blockDim.x + threadIdx.x;   // one float4 each
  v4f v = ((const v4f*)x)[i];
  v4h h = { (_Float16)v[0], (_Float16)v[1], (_Float16)v[2], (_Float16)v[3] };
  ((v4h*)xh)[i] = h;
}

// ------- kernel 2: dequant 4-bit + fold LoRA into W_eff (f16, [N][K]) --------
__global__ __launch_bounds__(256)
void k_build_weff(const int* __restrict__ qw, const __half* __restrict__ scales,
                  const __half* __restrict__ zeros, const float* __restrict__ lA,
                  const float* __restrict__ lB, _Float16* __restrict__ weff) {
  const int o = blockIdx.x;                       // one output row per block
  __shared__ float Brow[16];
  if (threadIdx.x < 16) Brow[threadIdx.x] = lB[o * 16 + threadIdx.x];
  __syncthreads();
  const float s = __half2float(scales[o]);
  const float z = __half2float(zeros[o]);
  for (int j = threadIdx.x; j < KDIM / 2; j += 256) {
    const int q  = qw[(size_t)o * (KDIM / 2) + j];
    const int i0 = 2 * j;                          // high nibble -> even col
    float w0 = (float)((q >> 4) & 15) * s + z;
    float w1 = (float)(q & 15) * s + z;
    float l0 = 0.f, l1 = 0.f;
#pragma unroll
    for (int r = 0; r < 16; ++r) {
      const float br = Brow[r];
      l0 += br * lA[(size_t)r * KDIM + i0];
      l1 += br * lA[(size_t)r * KDIM + i0 + 1];
    }
    v2h out = { (_Float16)(w0 + SCALING * l0), (_Float16)(w1 + SCALING * l1) };
    *(v2h*)&weff[(size_t)o * KDIM + i0] = out;
  }
}

// ---------------- kernel 3: WMMA GEMM  out = Xh @ Weff^T ---------------------
__global__ __launch_bounds__(256)
void k_gemm(const _Float16* __restrict__ Xh, const _Float16* __restrict__ Wf,
            float* __restrict__ out) {
  __shared__ __align__(16) _Float16 ldsX[2][BM][KPAD];
  __shared__ __align__(16) _Float16 ldsW[2][BN][KPAD];

  const int t     = threadIdx.x;
  const int lane  = t & 31;
  const int wave  = t >> 5;          // 8 waves: 2 (M) x 4 (N)
  const int mBase = blockIdx.y * BM;
  const int nBase = blockIdx.x * BN;
  const int wM    = (wave >> 2) * 64;
  const int wN    = (wave & 3) * 32;
  const int lrow  = lane & 15;
  const int hlf   = lane >> 4;       // half-wave select per CDNA5 WMMA layouts

  v8f acc[4][2] = {};

  // stage a 128x32 X tile + 128x32 W tile: 4 async b128 per thread per stage
  auto issue = [&](int buf, int k0) {
#pragma unroll
    for (int it = 0; it < 2; ++it) {
      const int c   = t + it * 256;       // 512 chunks of 16B per tile
      const int row = c >> 2;
      const int kc  = (c & 3) * 8;        // 8 halves per chunk
      g2l_16B(&Xh[(size_t)(mBase + row) * KDIM + k0 + kc], &ldsX[buf][row][kc]);
      g2l_16B(&Wf[(size_t)(nBase + row) * KDIM + k0 + kc], &ldsW[buf][row][kc]);
    }
  };

  const int nk = KDIM / BK;   // 128 k-steps
  issue(0, 0);
  for (int ks = 0; ks < nk; ++ks) {
    const int buf = ks & 1;
    if (ks + 1 < nk) {
      issue(buf ^ 1, (ks + 1) * BK);
      WAIT_ASYNC(4);                       // drain previous stage (4 ops/wave)
    } else {
      WAIT_ASYNC(0);
    }
    __syncthreads();                       // all waves' current tile visible

    // A frags 16x32 f16: lanes 0-15 row=lrow K[0..7]+[16..23]; lanes 16-31 +8
    v16h Af[4];
#pragma unroll
    for (int mf = 0; mf < 4; ++mf) {
      const _Float16* p = &ldsX[buf][wM + mf * 16 + lrow][hlf * 8];
      v8h lo = *(const v8h*)p;
      v8h hi = *(const v8h*)(p + 16);
      Af[mf] = __builtin_shufflevector(lo, hi, 0,1,2,3,4,5,6,7,8,9,10,11,12,13,14,15);
    }
    // B frags 32x16 f16: lane holds column n=lrow, K[0..7]+[16..23]; lanes 16-31 +8
    v16h Bf[2];
#pragma unroll
    for (int nf = 0; nf < 2; ++nf) {
      const _Float16* p = &ldsW[buf][wN + nf * 16 + lrow][hlf * 8];
      v8h lo = *(const v8h*)p;
      v8h hi = *(const v8h*)(p + 16);
      Bf[nf] = __builtin_shufflevector(lo, hi, 0,1,2,3,4,5,6,7,8,9,10,11,12,13,14,15);
    }
#pragma unroll
    for (int mf = 0; mf < 4; ++mf)
#pragma unroll
      for (int nf = 0; nf < 2; ++nf)
        acc[mf][nf] = __builtin_amdgcn_wmma_f32_16x16x32_f16(
            false, Af[mf], false, Bf[nf], (short)0, acc[mf][nf], false, false);
    __syncthreads();                       // protect buf before it is refilled
  }

  // C/D layout: VGPR r -> M = r + 8*hlf, N = lrow
#pragma unroll
  for (int mf = 0; mf < 4; ++mf)
#pragma unroll
    for (int nf = 0; nf < 2; ++nf) {
      float* po = &out[(size_t)(mBase + wM + mf * 16 + hlf * 8) * NDIM
                       + nBase + wN + nf * 16 + lrow];
#pragma unroll
      for (int r = 0; r < 8; ++r) po[(size_t)r * NDIM] = acc[mf][nf][r];
    }
}

// -----------------------------------------------------------------------------
extern "C" void kernel_launch(void* const* d_in, const int* in_sizes, int n_in,
                              void* d_out, int out_size, void* d_ws, size_t ws_size,
                              hipStream_t stream) {
  const float*  x      = (const float*)d_in[0];
  const int*    qw     = (const int*)d_in[1];
  const __half* scales = (const __half*)d_in[2];
  const __half* zeros  = (const __half*)d_in[3];
  const float*  lA     = (const float*)d_in[4];
  const float*  lB     = (const float*)d_in[5];
  float*        out    = (float*)d_out;

  _Float16* Xh = (_Float16*)d_ws;                                   // 64 MB
  _Float16* Wf = (_Float16*)((char*)d_ws + (size_t)MDIM * KDIM * 2); // +32 MB

  k_cvt_x<<<(size_t)MDIM * KDIM / 4 / 256, 256, 0, stream>>>(x, Xh);
  k_build_weff<<<NDIM, 256, 0, stream>>>(qw, scales, zeros, lA, lB, Wf);
  dim3 grid(NDIM / BN, MDIM / BM);
  k_gemm<<<grid, 256, 0, stream>>>(Xh, Wf, out);
}
